// FusedUpQnA1d_87282325389586
// MI455X (gfx1250) — matrix operand order repaired and checked by reading, not compiled, not run on hardware
//
#include <hip/hip_runtime.h>

// ---------------------------------------------------------------------------
// FusedUpQnA1d for MI455X (gfx1250, wave32, WMMA f16 split-precision).
//
//   prep_kernel   : qhat -> qWk split to f16 hi/lo [16][512], qb[16], rpe_exp
//   split_weights : Wv / Wo -> f16 hi/lo pairs (Markidis split, ~fp32 accuracy)
//   fused_kernel  : per (batch, 32-output window):
//     P1 x tile -> LDS transposed f16 hi/lo  [80 cols][512 K]
//     P2 qk = qWk^T*x via V_WMMA_F32_16X16X32_F16 (3 wmma / 32-K), exp,
//        normalized window weights
//     P3 v = Wv*x + bv via split-f16 WMMA -> LDS (f32)
//     P4 out[c][sf*32+wo] = sum_k a_k v[c][2wo-1+k] -> LDS f16 hi/lo (overlay)
//     P5 final = Wo*out + bo via split-f16 WMMA -> global
// ---------------------------------------------------------------------------

typedef _Float16 f16;
typedef __attribute__((ext_vector_type(8)))  _Float16 v8h;
typedef __attribute__((ext_vector_type(16))) _Float16 v16h;
typedef __attribute__((ext_vector_type(8)))  float    v8f;

#define NB    4
#define DIN   512
#define HID   512
#define HEADS 8
#define SF    2
#define HD    64
#define W_    8192
#define WOUT  4096

#define TWO   32          // output cols per tile (per sf)
#define TWIN  66          // input cols: 2*TWO + 2 (stride2, K4, pad1)
#define XCOLS 80          // padded to 5 WMMA N-tiles
#define XTS   520         // halves stride of transposed tiles (512 K + 8 pad)
#define VSTR  68          // f32 stride of v tile

// LDS byte offsets (total 318208 <= 320 KB)
#define XTH_OFF 0                         // [80][520] f16 = 83200
#define XTL_OFF 83200                     // [80][520] f16 = 83200
#define VS_OFF  166400                    // [512][68] f32 = 139264
#define CE_OFF  305664                    // [16][68]  f32 = 4352
#define AW_OFF  310016                    // [16][32][4] f32 = 8192
#define SMEM_BYTES 318208
// out tile overlays x region: hi at XTH_OFF, lo at XTH_OFF + 64*520*2 (66560)
#define OTH_OFF 0
#define OTL_OFF 66560

// workspace byte offsets
#define WSB_QB      0          // 16 f32
#define WSB_RPE     64         // 64 f32
#define WSB_QWKT_HI 512        // [16][512] f16 = 16384
#define WSB_QWKT_LO 16896
#define WSB_WV_HI   36864      // [512][512] f16 = 524288
#define WSB_WV_LO   561152
#define WSB_WO_HI   1085440
#define WSB_WO_LO   1609728    // end 2134016 (~2.04 MB workspace)

__device__ __forceinline__ v8f wmma16(v16h a, v16h b, v8f c) {
  return __builtin_amdgcn_wmma_f32_16x16x32_f16(false, a, false, b,
                                                (short)0, c, false, false);
}
// A operand: 16 contiguous halves
__device__ __forceinline__ v16h load_a16(const f16* p) {
  v16h r;
  *(v8h*)&r       = *(const v8h*)p;
  *((v8h*)&r + 1) = *(const v8h*)(p + 8);
  return r;
}
// B operand: 16 contiguous halves
__device__ __forceinline__ v16h load_b16(const f16* p) {
  v16h r;
  *(v8h*)&r       = *(const v8h*)p;
  *((v8h*)&r + 1) = *(const v8h*)(p + 8);
  return r;
}

// ---------------------------------------------------------------------------
__global__ __launch_bounds__(256)
void split_weights(const float* __restrict__ w, f16* __restrict__ hi,
                   f16* __restrict__ lo, int n) {
  for (int i = blockIdx.x * 256 + threadIdx.x; i < n; i += gridDim.x * 256) {
    float v = w[i];
    f16 h = (f16)v;
    hi[i] = h;
    lo[i] = (f16)(v - (float)h);
  }
}

// ---------------------------------------------------------------------------
__global__ __launch_bounds__(256)
void prep_kernel(const float* __restrict__ query,
                 const float* __restrict__ Wk,
                 const float* __restrict__ Wkb,
                 const float* __restrict__ rpe,
                 char* __restrict__ wsb) {
  __shared__ float qhat[SF * HID];
  const int tid = threadIdx.x;
  float* qb   = (float*)(wsb + WSB_QB);
  float* rpee = (float*)(wsb + WSB_RPE);
  f16* qth = (f16*)(wsb + WSB_QWKT_HI);
  f16* qtl = (f16*)(wsb + WSB_QWKT_LO);

  if (tid < SF * HEADS) {
    float ss = 0.f;
    for (int d = 0; d < HD; ++d) { float v = query[tid * HD + d]; ss += v * v; }
    float inv = 1.0f / ((sqrtf(ss) + 1e-6f) * 8.0f);   // /(norm+eps)/sqrt(HD)
    for (int d = 0; d < HD; ++d) qhat[tid * HD + d] = query[tid * HD + d] * inv;
  }
  __syncthreads();

  // qWk[g][D] = sum_d qhat[g][d] * Wk[D][h*64+d], g = sf*8+h; split to hi/lo
  for (int i = tid; i < DIN * 16; i += 256) {
    int D = i >> 4, g = i & 15, h = g & 7;
    const float* wkrow = Wk + (size_t)D * HID + h * HD;
    const float* qh = qhat + g * HD;
    float acc = 0.f;
    #pragma unroll 8
    for (int d = 0; d < HD; ++d) acc += qh[d] * wkrow[d];
    f16 hv = (f16)acc;
    qth[g * DIN + D] = hv;
    qtl[g * DIN + D] = (f16)(acc - (float)hv);
  }

  if (tid < SF * HEADS) {
    int h = tid & 7;
    float acc = 0.f;
    for (int d = 0; d < HD; ++d) acc += qhat[tid * HD + d] * Wkb[h * HD + d];
    qb[tid] = acc;
  }
  if (tid < SF * HEADS * 4) rpee[tid] = expf(rpe[tid]);
}

// ---------------------------------------------------------------------------
__global__ __launch_bounds__(256)
void fused_kernel(const float* __restrict__ x,
                  const float* __restrict__ bv,
                  const float* __restrict__ bo,
                  const char* __restrict__ wsb,
                  float* __restrict__ out) {
  extern __shared__ char smem[];
  f16*   xth  = (f16*)(smem + XTH_OFF);
  f16*   xtl  = (f16*)(smem + XTL_OFF);
  float* vs   = (float*)(smem + VS_OFF);
  float* ce   = (float*)(smem + CE_OFF);
  float* aw   = (float*)(smem + AW_OFF);
  f16*   oth  = (f16*)(smem + OTH_OFF);   // overlays x tile (dead after P3)
  f16*   otl  = (f16*)(smem + OTL_OFF);

  const int tid  = threadIdx.x;
  const int lane = tid & 31;
  const int wave = tid >> 5;
  const int b    = blockIdx.y;
  const int o0   = blockIdx.x * TWO;
  const int wbase = 2 * o0 - 1;

  const int lcol  = lane & 15;
  const int asub  = (lane >> 4) << 4;   // A K sub-offset: 0 | 16
  const int bsub  = (lane >> 4) << 4;   // B K sub-offset: 0 | 16
  const int rhalf = (lane >> 4) << 3;   // C row sub-offset: 0 | 8

  // ---- P1: x tile -> LDS, transposed, split f16 hi/lo ----
  const float* xb = x + (size_t)b * DIN * W_;
  for (int i = tid; i < DIN * XCOLS; i += 256) {
    int d = i / XCOLS, wl = i - d * XCOLS;
    int w = wbase + wl;
    float v = 0.f;
    if (wl < TWIN && w >= 0 && w < W_) v = xb[(size_t)d * W_ + w];
    f16 h = (f16)v;
    xth[wl * XTS + d] = h;
    xtl[wl * XTS + d] = (f16)(v - (float)h);
  }
  __syncthreads();

  // ---- P2a: qk = qWk^T * x (waves 0..4, one 16-col N tile each) ----
  if (wave < 5) {
    const int col = wave * 16 + lcol;
    const f16* qh = (const f16*)(wsb + WSB_QWKT_HI) + lcol * DIN;
    const f16* ql = (const f16*)(wsb + WSB_QWKT_LO) + lcol * DIN;
    const f16* bhp = xth + col * XTS;
    const f16* blp = xtl + col * XTS;
    v8f acc = {};
    #pragma unroll 2
    for (int k0 = 0; k0 < DIN; k0 += 32) {
      v16h ah = load_a16(qh + k0 + asub);
      v16h al = load_a16(ql + k0 + asub);
      v16h bh = load_b16(bhp + k0 + bsub);
      v16h bl = load_b16(blp + k0 + bsub);
      acc = wmma16(ah, bh, acc);
      acc = wmma16(al, bh, acc);
      acc = wmma16(ah, bl, acc);
    }
    const float* qb = (const float*)(wsb + WSB_QB);
    const int w0 = wbase + col;
    if (col < TWIN) {
      const bool inr = (w0 >= 0) && (w0 < W_);
      #pragma unroll
      for (int v = 0; v < 8; ++v) {
        int g = v + rhalf;
        float e = expf(acc[v] + qb[g]);
        ce[g * 68 + col] = inr ? e : 0.f;   // zero outside conv padding
      }
    }
  }
  __syncthreads();

  // ---- P2b: normalized window weights a[g][wo][k] ----
  for (int it = tid; it < SF * HEADS * TWO; it += 256) {
    int wol = it & 31, g = it >> 5;
    const float* rp = (const float*)(wsb + WSB_RPE) + g * 4;
    float a0 = rp[0] * ce[g * 68 + 2 * wol + 0];
    float a1 = rp[1] * ce[g * 68 + 2 * wol + 1];
    float a2 = rp[2] * ce[g * 68 + 2 * wol + 2];
    float a3 = rp[3] * ce[g * 68 + 2 * wol + 3];
    float inv = 1.0f / (a0 + a1 + a2 + a3);
    float* dst = aw + (g * 32 + wol) * 4;
    dst[0] = a0 * inv; dst[1] = a1 * inv; dst[2] = a2 * inv; dst[3] = a3 * inv;
  }

  // ---- P3: v = Wv * x + bv (32 M-tiles x 5 N-tiles over 8 waves) ----
  for (int t = wave; t < 32 * 5; t += 8) {
    const int m0 = (t / 5) * 16;
    const int col = (t % 5) * 16 + lcol;
    const int rbase = m0 + rhalf;
    const f16* wh = (const f16*)(wsb + WSB_WV_HI) + (size_t)(m0 + lcol) * DIN;
    const f16* wl = (const f16*)(wsb + WSB_WV_LO) + (size_t)(m0 + lcol) * DIN;
    const f16* bhp = xth + col * XTS;
    const f16* blp = xtl + col * XTS;
    v8f acc;
    #pragma unroll
    for (int v = 0; v < 8; ++v) acc[v] = bv[rbase + v];
    #pragma unroll 2
    for (int k0 = 0; k0 < DIN; k0 += 32) {
      v16h ah = load_a16(wh + k0 + asub);
      v16h al = load_a16(wl + k0 + asub);
      v16h bh = load_b16(bhp + k0 + bsub);
      v16h bl = load_b16(blp + k0 + bsub);
      acc = wmma16(ah, bh, acc);
      acc = wmma16(al, bh, acc);
      acc = wmma16(ah, bl, acc);
    }
    if (col < TWIN) {
      #pragma unroll
      for (int v = 0; v < 8; ++v) vs[(rbase + v) * VSTR + col] = acc[v];
    }
  }
  __syncthreads();

  // ---- P4: out[c][j] = sum_k a_k * v[c][2wo+k]; split f16 -> overlay ----
  for (int i = tid; i < HID * 64; i += 256) {
    int c = i >> 6, j = i & 63;
    int sf = j >> 5, wol = j & 31;
    int g = sf * 8 + (c >> 6);
    const float* awp = aw + (g * 32 + wol) * 4;
    const float* vp  = vs + c * VSTR + 2 * wol;
    float o = awp[0] * vp[0] + awp[1] * vp[1] + awp[2] * vp[2] + awp[3] * vp[3];
    f16 h = (f16)o;
    oth[j * XTS + c] = h;
    otl[j * XTS + c] = (f16)(o - (float)h);
  }
  __syncthreads();

  // ---- P5: final = Wo * out + bo (32 M-tiles x 4 N-tiles) -> global ----
  float* ob = out + (size_t)b * HID * (SF * WOUT);
  for (int t = wave; t < 32 * 4; t += 8) {
    const int m0 = (t >> 2) * 16;
    const int col = (t & 3) * 16 + lcol;   // j in [0,64)
    const int rbase = m0 + rhalf;
    const f16* wh = (const f16*)(wsb + WSB_WO_HI) + (size_t)(m0 + lcol) * HID;
    const f16* wl = (const f16*)(wsb + WSB_WO_LO) + (size_t)(m0 + lcol) * HID;
    const f16* bhp = oth + col * XTS;
    const f16* blp = otl + col * XTS;
    v8f acc;
    #pragma unroll
    for (int v = 0; v < 8; ++v) acc[v] = bo[rbase + v];
    #pragma unroll 2
    for (int k0 = 0; k0 < HID; k0 += 32) {
      v16h ah = load_a16(wh + k0 + asub);
      v16h al = load_a16(wl + k0 + asub);
      v16h bh = load_b16(bhp + k0 + bsub);
      v16h bl = load_b16(blp + k0 + bsub);
      acc = wmma16(ah, bh, acc);
      acc = wmma16(al, bh, acc);
      acc = wmma16(ah, bl, acc);
    }
    const int sf = col >> 5, wol = col & 31;
    const size_t wpos = (size_t)sf * WOUT + o0 + wol;
    #pragma unroll
    for (int v = 0; v < 8; ++v)
      ob[(size_t)(rbase + v) * (SF * WOUT) + wpos] = acc[v];
  }
}

// ---------------------------------------------------------------------------
extern "C" void kernel_launch(void* const* d_in, const int* in_sizes, int n_in,
                              void* d_out, int out_size, void* d_ws, size_t ws_size,
                              hipStream_t stream) {
  (void)in_sizes; (void)n_in; (void)out_size; (void)ws_size;
  const float* x     = (const float*)d_in[0];
  const float* query = (const float*)d_in[1];
  const float* Wk    = (const float*)d_in[2];
  const float* Wkb   = (const float*)d_in[3];
  const float* rpe   = (const float*)d_in[4];
  const float* Wv    = (const float*)d_in[5];
  const float* bv    = (const float*)d_in[6];
  const float* Wo    = (const float*)d_in[7];
  const float* bo    = (const float*)d_in[8];
  char*  wsb = (char*)d_ws;
  float* out = (float*)d_out;

  prep_kernel<<<1, 256, 0, stream>>>(query, Wk, Wkb, rpe, wsb);
  split_weights<<<256, 256, 0, stream>>>(Wv, (f16*)(wsb + WSB_WV_HI),
                                         (f16*)(wsb + WSB_WV_LO), HID * DIN);
  split_weights<<<256, 256, 0, stream>>>(Wo, (f16*)(wsb + WSB_WO_HI),
                                         (f16*)(wsb + WSB_WO_LO), HID * HID);

  dim3 grid(WOUT / TWO, NB);   // 128 x 4 workgroups
  fused_kernel<<<grid, 256, SMEM_BYTES, stream>>>(x, bv, bo, wsb, out);
}
